// SoftAlignmentModule_4063039062717
// MI455X (gfx1250) — matrix operand order repaired and checked by reading, not compiled, not run on hardware
//
#include <hip/hip_runtime.h>
#include <hip/hip_bf16.h>

typedef __attribute__((ext_vector_type(16))) __bf16       v16bf;
typedef __attribute__((ext_vector_type(8)))  float        v8f;
typedef __attribute__((ext_vector_type(4)))  unsigned int v4u;

union Frag {
    v4u   u[2];
    v16bf bf;
};

// Raw max (skip LLVM's NaN canonicalization; inputs are real data)
static __device__ __forceinline__ float vmaxf(float a, float b) {
    float d;
    asm("v_max_num_f32_e32 %0, %1, %2" : "=v"(d) : "v"(a), "v"(b));
    return d;
}

// Max-reduce across the 16-lane DPP row, one fused v_max_num_f32_dpp per step.
// Result broadcast to all 16 lanes of the row.  Pure VALU, no LDS traffic.
static __device__ __forceinline__ float red_max16(float x) {
    float t;
    asm("v_max_num_f32_dpp %0, %1, %1 quad_perm:[1,0,3,2] row_mask:0xf bank_mask:0xf bound_ctrl:1"
        : "=v"(t) : "v"(x));
    asm("v_max_num_f32_dpp %0, %1, %1 quad_perm:[2,3,0,1] row_mask:0xf bank_mask:0xf bound_ctrl:1"
        : "=v"(x) : "v"(t));
    asm("v_max_num_f32_dpp %0, %1, %1 row_half_mirror row_mask:0xf bank_mask:0xf bound_ctrl:1"
        : "=v"(t) : "v"(x));
    asm("v_max_num_f32_dpp %0, %1, %1 row_mirror row_mask:0xf bank_mask:0xf bound_ctrl:1"
        : "=v"(x) : "v"(t));
    return x;
}

static __device__ __forceinline__ __bf16 f2bf(float f) { return (__bf16)f; }

#define LOG2E 1.4426950408889634f

// ---------------------------------------------------------------------------
// Kernel 1: shared 1x1-conv projection.  q/p stored n-major [B][4096][32] bf16
// so WMMA A/B fragments are contiguous b128 loads.  q is pre-scaled by log2e
// so the attention softmax can use raw v_exp_f32 (exp2).
// ---------------------------------------------------------------------------
__global__ __launch_bounds__(256) void proj_kernel(
    const float* __restrict__ Fq, const float* __restrict__ Fp,
    const float* __restrict__ W,  const float* __restrict__ bias,
    __bf16* __restrict__ qo, __bf16* __restrict__ po)
{
    const int n = blockIdx.x * blockDim.x + threadIdx.x; // 0..4095
    const int o = blockIdx.y;                            // 0..31
    const int b = blockIdx.z;                            // 0..3
    const float* fq = Fq + (size_t)b * 64 * 4096 + n;
    const float* fp = Fp + (size_t)b * 64 * 4096 + n;
    const float* w  = W + o * 64;
    float sq = bias[o], sp = bias[o];
#pragma unroll 8
    for (int c = 0; c < 64; ++c) {
        float wc = w[c];
        sq = fmaf(wc, fq[(size_t)c * 4096], sq);
        sp = fmaf(wc, fp[(size_t)c * 4096], sp);
    }
    size_t idx = ((size_t)b * 4096 + n) * 32 + o;
    qo[idx] = f2bf(sq * LOG2E);   // exp2-domain for softmax
    po[idx] = f2bf(sp);
}

// ---------------------------------------------------------------------------
// Kernel 2: Fp (f32, c-major) -> v (bf16, c-major [B][64][4096])
// ---------------------------------------------------------------------------
__global__ __launch_bounds__(256) void cvt_kernel(
    const float* __restrict__ Fp, __bf16* __restrict__ vo)
{
    int t = blockIdx.x * blockDim.x + threadIdx.x;  // 0..262143, 4 elems each
    const float4* in = (const float4*)Fp;
    float4 x = in[t];
    __bf16* o = vo + (size_t)t * 4;
    o[0] = f2bf(x.x); o[1] = f2bf(x.y); o[2] = f2bf(x.z); o[3] = f2bf(x.w);
}

// ---------------------------------------------------------------------------
// Kernel 3: flash-attention over HW=4096 positions, head dim 32, value dim 64.
// One wave owns one 16-row output tile (16x64 f32 accum = 4 C-tiles).
// Per 32-column key chunk: 2 WMMAs for S (p-fragments software-pipelined one
// iteration ahead), online softmax in the exp2 domain (fused DPP row-max, the
// denominator tracked by a 5th WMMA against an all-ones B matrix), P
// transposed through per-wave LDS, 4 WMMAs accumulating O against v.
// ---------------------------------------------------------------------------
__global__ __launch_bounds__(256) void attn_kernel(
    const __bf16* __restrict__ q, const __bf16* __restrict__ p,
    const __bf16* __restrict__ v, float* __restrict__ out)
{
    __shared__ __align__(16) __bf16 ldsP[8][512];   // per-wave 16x32 bf16 P tile

    const int wave = threadIdx.x >> 5;
    const int lane = threadIdx.x & 31;
    const int half = lane >> 4;      // 0: lanes 0-15, 1: lanes 16-31
    const int l15  = lane & 15;

    const int tile = blockIdx.x * 8 + wave;   // 0..1023 (256 tiles per batch)
    const int b    = tile >> 8;
    const int n0   = (tile & 255) << 4;

    const __bf16* qb = q + (size_t)b * 4096 * 32;
    const __bf16* pb = p + (size_t)b * 4096 * 32;
    const __bf16* vb = v + (size_t)b * 64 * 4096;

    // A-fragment of q (16x32, contiguous k = 16*half + s order):
    // lanes 0-15: K 0..15; lanes 16-31: K 16..31
    Frag aq;
    {
        const v4u* r = (const v4u*)(qb + (size_t)(n0 + l15) * 32);
        aq.u[0] = r[2 * half];
        aq.u[1] = r[2 * half + 1];
    }

    // All-ones 32x16 bf16 B fragment: P x ones accumulates the softmax
    // denominator into every column of acc4.
    Frag ones;
    ones.u[0] = (v4u){0x3F803F80u, 0x3F803F80u, 0x3F803F80u, 0x3F803F80u};
    ones.u[1] = (v4u){0x3F803F80u, 0x3F803F80u, 0x3F803F80u, 0x3F803F80u};

    v8f acc0 = {}, acc1 = {}, acc2 = {}, acc3 = {}, acc4 = {};
    float mrow[8];
#pragma unroll
    for (int i = 0; i < 8; ++i) mrow[i] = -3.0e38f;

    __bf16* lp = &ldsP[wave][0];

    // B-fragments of p (32x16, same contiguous k order as aq):
    // lanes 0-15 hold K=0..15, lanes 16-31 hold K=16..31
    Frag pB0, pB1, pB0n, pB1n;
    {
        const v4u* r0 = (const v4u*)(pb + (size_t)l15 * 32);
        pB0.u[0] = r0[2 * half]; pB0.u[1] = r0[2 * half + 1];
        const v4u* r1 = (const v4u*)(pb + (size_t)(16 + l15) * 32);
        pB1.u[0] = r1[2 * half]; pB1.u[1] = r1[2 * half + 1];
    }

    for (int m0 = 0; m0 < 4096; m0 += 32) {
        // Prefetch next chunk's p fragments (wrap-around keeps EXEC uniform).
        {
            int mn_ = (m0 + 32) & 4095;
            const v4u* r0 = (const v4u*)(pb + (size_t)(mn_ + l15) * 32);
            pB0n.u[0] = r0[2 * half]; pB0n.u[1] = r0[2 * half + 1];
            const v4u* r1 = (const v4u*)(pb + (size_t)(mn_ + 16 + l15) * 32);
            pB1n.u[0] = r1[2 * half]; pB1n.u[1] = r1[2 * half + 1];
        }

        v8f z = {};
        v8f S0 = __builtin_amdgcn_wmma_f32_16x16x32_bf16(false, aq.bf, false, pB0.bf,
                                                         (short)0, z, false, false);
        v8f S1 = __builtin_amdgcn_wmma_f32_16x16x32_bf16(false, aq.bf, false, pB1.bf,
                                                         (short)0, z, false, false);

        // Online softmax (exp2 domain). C-layout: VGPR i <-> row i+8*half, col l15.
#pragma unroll
        for (int i = 0; i < 8; ++i) {
            float t  = red_max16(vmaxf(S0[i], S1[i]));
            float mn = vmaxf(mrow[i], t);
            float sc = __builtin_amdgcn_exp2f(mrow[i] - mn);
            mrow[i] = mn;
            float p0 = __builtin_amdgcn_exp2f(S0[i] - mn);
            float p1 = __builtin_amdgcn_exp2f(S1[i] - mn);
            acc0[i] *= sc; acc1[i] *= sc; acc2[i] *= sc; acc3[i] *= sc; acc4[i] *= sc;
            int r = i + 8 * half;
            lp[r * 32 + l15]      = f2bf(p0);   // k = l15       (tile m0)
            lp[r * 32 + l15 + 16] = f2bf(p1);   // k = l15 + 16  (tile m0+16)
        }

        // Reload P as a 16x32 A-fragment (wave-private LDS, DS ops in-order).
        Frag pa;
        {
            const v4u* r = (const v4u*)(lp + l15 * 32);
            pa.u[0] = r[2 * half];
            pa.u[1] = r[2 * half + 1];
        }

        // Softmax denominator: acc4 += P * ones
        acc4 = __builtin_amdgcn_wmma_f32_16x16x32_bf16(false, pa.bf, false, ones.bf,
                                                       (short)0, acc4, false, false);

        // O[.,cg] += P * v-slice  (B-fragment of v: K along n, contiguous)
#pragma unroll
        for (int cg = 0; cg < 4; ++cg) {
            Frag vB;
            const v4u* r = (const v4u*)(vb + (size_t)(cg * 16 + l15) * 4096 + m0);
            vB.u[0] = r[2 * half]; vB.u[1] = r[2 * half + 1];
            v8f* a = (cg == 0) ? &acc0 : (cg == 1) ? &acc1 : (cg == 2) ? &acc2 : &acc3;
            *a = __builtin_amdgcn_wmma_f32_16x16x32_bf16(false, pa.bf, false, vB.bf,
                                                         (short)0, *a, false, false);
        }

        pB0 = pB0n; pB1 = pB1n;
    }

    // Every lane already holds its rows' denominators in acc4 (all columns equal).
    float inv[8];
#pragma unroll
    for (int i = 0; i < 8; ++i) inv[i] = __builtin_amdgcn_rcpf(acc4[i]);

    // out[b][c][n] f32
#pragma unroll
    for (int cg = 0; cg < 4; ++cg) {
        v8f a = (cg == 0) ? acc0 : (cg == 1) ? acc1 : (cg == 2) ? acc2 : acc3;
        float* op = out + ((size_t)b * 64 + cg * 16 + l15) * 4096 + n0 + 8 * half;
#pragma unroll
        for (int i = 0; i < 8; ++i) op[i] = a[i] * inv[i];
    }
}

extern "C" void kernel_launch(void* const* d_in, const int* in_sizes, int n_in,
                              void* d_out, int out_size, void* d_ws, size_t ws_size,
                              hipStream_t stream) {
    const float* Fq   = (const float*)d_in[0];
    const float* Fp   = (const float*)d_in[1];
    const float* W    = (const float*)d_in[2];
    const float* bias = (const float*)d_in[3];
    float* out = (float*)d_out;

    char* ws = (char*)d_ws;
    const size_t QP = (size_t)4 * 4096 * 32 * 2;   // 1 MB each
    __bf16* qws = (__bf16*)(ws);
    __bf16* pws = (__bf16*)(ws + QP);
    __bf16* vws = (__bf16*)(ws + 2 * QP);          // 2 MB

    dim3 pg(16, 32, 4);
    proj_kernel<<<pg, 256, 0, stream>>>(Fq, Fp, W, bias, qws, pws);
    cvt_kernel<<<1024, 256, 0, stream>>>(Fp, vws);
    attn_kernel<<<128, 256, 0, stream>>>(qws, pws, vws, out);
}
